// TargetModel_77223511982363
// MI455X (gfx1250) — matrix operand order repaired and checked by reading, not compiled, not run on hardware
//
#include <hip/hip_runtime.h>
#include <hip/hip_bf16.h>
#include <math.h>

// ---- problem constants (match reference) ----
#define NEG_SLOPE 0.2f
#define LN_EPS    1e-5f
#define BB     4
#define NN     10000
#define DD     128
#define HIDD   256
#define HEADS_ 2
#define FHEAD  128   // HIDD / HEADS_

typedef float    v2f  __attribute__((ext_vector_type(2)));
typedef float    v8f  __attribute__((ext_vector_type(8)));
typedef _Float16 v16h __attribute__((ext_vector_type(16)));

// order-preserving float<->uint encoding for atomicMax on f32
__device__ __forceinline__ unsigned f2ord(float f) {
  unsigned b = __float_as_uint(f);
  return (b & 0x80000000u) ? ~b : (b | 0x80000000u);
}
__device__ __forceinline__ float ord2f(unsigned u) {
  return (u & 0x80000000u) ? __uint_as_float(u & 0x7FFFFFFFu)
                           : __uint_as_float(~u);
}

// ---------------------------------------------------------------------------
// GEMM: H[b] = X[b] @ W   (X: [N,K] per graph, W: [K,HIDD], H: [N,HIDD])
// One wave computes one 16x16 output tile, K stepped per the WMMA shape.
// Tile guard is wave-uniform -> EXEC all-1s inside WMMA (ISA requirement).
// ---------------------------------------------------------------------------
__global__ __launch_bounds__(256) void gemm_wmma(
    const float* __restrict__ X, const float* __restrict__ W,
    float* __restrict__ Hout, int K, int totalTiles, int nodeTiles, int colTiles)
{
  const int wave = threadIdx.x >> 5;
  const int lane = threadIdx.x & 31;
  const int tile = blockIdx.x * 8 + wave;
  if (tile >= totalTiles) return;                 // wave-uniform

  const int tilesPerBatch = nodeTiles * colTiles;
  const int b   = tile / tilesPerBatch;
  const int rem = tile % tilesPerBatch;
  const int nt  = rem / colTiles;
  const int ct  = rem % colTiles;

  const float* Xb = X + (size_t)b * NN * K;
  float*       Hb = Hout + (size_t)b * NN * HIDD;
  const int row0 = nt * 16;
  const int col0 = ct * 16;
  const int half = lane >> 4;   // 0: lanes 0-15, 1: lanes 16-31
  const int r    = lane & 15;

  const float* xrow = Xb + (size_t)(row0 + r) * K;
  v8f acc = {0.f, 0.f, 0.f, 0.f, 0.f, 0.f, 0.f, 0.f};

#if __has_builtin(__builtin_amdgcn_wmma_f32_16x16x4_f32)
  // Full-precision f32 WMMA. A 16x4 layout: v0 = K0(K2), v1 = K1(K3) per half.
  // B 4x16: rows striped across lanes; v0 = K0(K2), v1 = K1(K3).
  for (int k0 = 0; k0 < K; k0 += 4) {
    v2f a, bv;
    a.x  = xrow[k0 + 2 * half];
    a.y  = xrow[k0 + 2 * half + 1];
    bv.x = W[(size_t)(k0 + 2 * half)     * HIDD + col0 + r];
    bv.y = W[(size_t)(k0 + 2 * half + 1) * HIDD + col0 + r];
    acc = __builtin_amdgcn_wmma_f32_16x16x4_f32(
        /*neg_a=*/false, a, /*neg_b=*/false, bv,
        /*c_mod=*/(short)0, acc, /*reuse_a=*/false, /*reuse_b=*/false);
  }
#else
  // Fallback: codegen-confirmed f16 WMMA, convert on the fly.
  for (int k0 = 0; k0 < K; k0 += 32) {
    v16h av, bv;
#pragma unroll
    for (int j = 0; j < 8; ++j) {
      // A 16x32 f16 layout (ISA 7.12.2): v0-3 -> K 0..7 (+8 for upper half),
      // v4-7 -> K 16..23 (+8 for upper half); packed lo/hi per VGPR.
      int ka = (j < 4) ? (2 * j + (half ? 8 : 0))
                       : (16 + 2 * (j - 4) + (half ? 8 : 0));
      av[2 * j]     = (_Float16)xrow[k0 + ka];
      av[2 * j + 1] = (_Float16)xrow[k0 + ka + 1];
      // B 32x16: same K mapping as A (element 2j/2j+1 -> K=ka,ka+1)
      int kb = ka;
      bv[2 * j]     = (_Float16)W[(size_t)(k0 + kb)     * HIDD + col0 + r];
      bv[2 * j + 1] = (_Float16)W[(size_t)(k0 + kb + 1) * HIDD + col0 + r];
    }
    acc = __builtin_amdgcn_wmma_f32_16x16x32_f16(
        false, av, false, bv, (short)0, acc, false, false);
  }
#endif

  // C/D layout: VGPR i -> M = i (lanes 0-15) / i+8 (lanes 16-31), N = lane%16
#pragma unroll
  for (int i = 0; i < 8; ++i) {
    int m = i + 8 * half;
    Hb[(size_t)(row0 + m) * HIDD + col0 + r] = acc[i];
  }
}

// ---------------------------------------------------------------------------
// Per-(b,node,head) attention logits: al_s = <h, a_src>, al_d = <h, a_dst>
// One wave per (b*N, head); 128 features / 32 lanes = 4 each + shfl reduce.
// ---------------------------------------------------------------------------
__global__ __launch_bounds__(256) void attn_logits(
    const float* __restrict__ H, const float* __restrict__ a_src,
    const float* __restrict__ a_dst, float* __restrict__ alS,
    float* __restrict__ alD, int totalWaves)
{
  int wave = (int)((blockIdx.x * (unsigned)blockDim.x + threadIdx.x) >> 5);
  int lane = threadIdx.x & 31;
  if (wave >= totalWaves) return;
  int h_ = wave % HEADS_;
  int bn = wave / HEADS_;  // b*N + n
  const float* hp = H + (size_t)bn * HIDD + h_ * FHEAD;
  const float* as = a_src + h_ * FHEAD;
  const float* ad = a_dst + h_ * FHEAD;
  float ss = 0.f, sd = 0.f;
#pragma unroll
  for (int j = 0; j < FHEAD / 32; ++j) {
    int f = lane + 32 * j;
    float v = hp[f];
    ss += v * as[f];
    sd += v * ad[f];
  }
  for (int off = 16; off; off >>= 1) {
    ss += __shfl_xor(ss, off, 32);
    sd += __shfl_xor(sd, off, 32);
  }
  if (lane == 0) { alS[wave] = ss; alD[wave] = sd; }
}

__global__ void init_seg(unsigned* __restrict__ m, float* __restrict__ denom,
                         int total) {
  int i = blockIdx.x * blockDim.x + threadIdx.x;
  if (i < total) { m[i] = 0x007FFFFFu; /* f2ord(-inf) */ denom[i] = 0.f; }
}

__global__ void zero_f4(float4* __restrict__ p, long count4) {
  long i = blockIdx.x * (long)blockDim.x + threadIdx.x;
  if (i < count4) p[i] = make_float4(0.f, 0.f, 0.f, 0.f);
}

__device__ __forceinline__ float edge_logit(const float* alS, const float* alD,
                                            int b, int s, int d, int h_) {
  float e = alS[(b * NN + s) * HEADS_ + h_] + alD[(b * NN + d) * HEADS_ + h_];
  return e > 0.f ? e : NEG_SLOPE * e;
}

__global__ void edge_max_k(const int* __restrict__ src, const int* __restrict__ dst,
                           const float* __restrict__ alS, const float* __restrict__ alD,
                           unsigned* __restrict__ mseg, int E, long total)
{
  long i = blockIdx.x * (long)blockDim.x + threadIdx.x;
  if (i >= total) return;
  int h_ = (int)(i % HEADS_);
  long rest = i / HEADS_;
  int e = (int)(rest % E);
  int b = (int)(rest / E);
  int s = src[e], d = dst[e];
  float ev = edge_logit(alS, alD, b, s, d, h_);
  atomicMax(&mseg[(b * NN + d) * HEADS_ + h_], f2ord(ev));
}

__global__ void edge_exp_k(const int* __restrict__ src, const int* __restrict__ dst,
                           const float* __restrict__ alS, const float* __restrict__ alD,
                           const unsigned* __restrict__ mseg, float* __restrict__ denom,
                           float* __restrict__ exbuf, int E, long total)
{
  long i = blockIdx.x * (long)blockDim.x + threadIdx.x;
  if (i >= total) return;
  int h_ = (int)(i % HEADS_);
  long rest = i / HEADS_;
  int e = (int)(rest % E);
  int b = (int)(rest / E);
  int s = src[e], d = dst[e];
  float ev = edge_logit(alS, alD, b, s, d, h_);
  float mx = ord2f(mseg[(b * NN + d) * HEADS_ + h_]);
  float ex = __expf(ev - mx);
  exbuf[i] = ex;
  atomicAdd(&denom[(b * NN + d) * HEADS_ + h_], ex);
}

// AGG[b,dst,h,f] += h[b,src,h,f] * alpha ; one thread per float4 chunk
__global__ void scatter_k(const int* __restrict__ src, const int* __restrict__ dst,
                          const float* __restrict__ H, const float* __restrict__ exbuf,
                          const float* __restrict__ denom, float* __restrict__ AGG,
                          int E, long total)
{
  long i = blockIdx.x * (long)blockDim.x + threadIdx.x;
  if (i >= total) return;
  const int FC = FHEAD / 4;          // 32 float4 chunks per head
  int fc = (int)(i % FC);
  long rest = i / FC;
  int h_ = (int)(rest % HEADS_);
  rest /= HEADS_;
  int e = (int)(rest % E);
  int b = (int)(rest / E);
  int s = src[e], d = dst[e];
  float ex = exbuf[((long)b * E + e) * HEADS_ + h_];
  float alpha = ex / (denom[(b * NN + d) * HEADS_ + h_] + 1e-16f);
  const float4 v = *(const float4*)&H[((size_t)(b * NN + s)) * HIDD + h_ * FHEAD + fc * 4];
  float* o = &AGG[((size_t)(b * NN + d)) * HIDD + h_ * FHEAD + fc * 4];
  atomicAdd(o + 0, v.x * alpha);
  atomicAdd(o + 1, v.y * alpha);
  atomicAdd(o + 2, v.z * alpha);
  atomicAdd(o + 3, v.w * alpha);
}

// out = relu(LN(AGG + bias) * g + be) ; one wave per (b,node)
__global__ __launch_bounds__(256) void ln_relu_k(
    const float* __restrict__ AGG, const float* __restrict__ bias,
    const float* __restrict__ g, const float* __restrict__ be,
    float* __restrict__ out, int totalNodes)
{
  int wave = (int)((blockIdx.x * (unsigned)blockDim.x + threadIdx.x) >> 5);
  int lane = threadIdx.x & 31;
  if (wave >= totalNodes) return;
  const float* x = AGG + (size_t)wave * HIDD;
  float v[HIDD / 32];
  float s = 0.f, s2 = 0.f;
#pragma unroll
  for (int j = 0; j < HIDD / 32; ++j) {
    int c = lane + 32 * j;
    float t = x[c] + bias[c];
    v[j] = t; s += t; s2 += t * t;
  }
  for (int off = 16; off; off >>= 1) {
    s  += __shfl_xor(s,  off, 32);
    s2 += __shfl_xor(s2, off, 32);
  }
  const float mu   = s * (1.f / HIDD);
  const float var  = s2 * (1.f / HIDD) - mu * mu;
  const float rstd = rsqrtf(var + LN_EPS);
  float* o = out + (size_t)wave * HIDD;
#pragma unroll
  for (int j = 0; j < HIDD / 32; ++j) {
    int c = lane + 32 * j;
    float t = (v[j] - mu) * rstd * g[c] + be[c];
    o[c] = t > 0.f ? t : 0.f;
  }
}

static inline int cdiv(long a, long b) { return (int)((a + b - 1) / b); }

extern "C" void kernel_launch(void* const* d_in, const int* in_sizes, int n_in,
                              void* d_out, int out_size, void* d_ws, size_t ws_size,
                              hipStream_t stream) {
  const float* feature = (const float*)d_in[0];
  const int*   eidx    = (const int*)d_in[1];
  const float* W1  = (const float*)d_in[2];
  const float* aS1 = (const float*)d_in[3];
  const float* aD1 = (const float*)d_in[4];
  const float* b1  = (const float*)d_in[5];
  const float* g1  = (const float*)d_in[6];
  const float* be1 = (const float*)d_in[7];
  const float* W2  = (const float*)d_in[8];
  const float* aS2 = (const float*)d_in[9];
  const float* aD2 = (const float*)d_in[10];
  const float* b2  = (const float*)d_in[11];
  const float* g2  = (const float*)d_in[12];
  const float* be2 = (const float*)d_in[13];

  const int E = in_sizes[1] / 2;       // edge_index is [2, E]
  const int* src = eidx;
  const int* dst = eidx + E;

  // workspace carve-up (~89 MB)
  float*    H     = (float*)d_ws;                          // [B,N,HID]
  float*    AGG   = H   + (size_t)BB * NN * HIDD;          // [B,N,HID]
  float*    alS   = AGG + (size_t)BB * NN * HIDD;          // [B,N,H]
  float*    alD   = alS + (size_t)BB * NN * HEADS_;        // [B,N,H]
  unsigned* mseg  = (unsigned*)(alD + (size_t)BB * NN * HEADS_);
  float*    denom = (float*)(mseg + (size_t)BB * NN * HEADS_);
  float*    exbuf = denom + (size_t)BB * NN * HEADS_;      // [B,E,H]

  const int  nodeTiles  = NN / 16;          // 625 (exact)
  const int  colTiles   = HIDD / 16;        // 16
  const int  totalTiles = BB * nodeTiles * colTiles;
  const int  gemmBlocks = cdiv(totalTiles, 8);
  const int  segTotal   = BB * NN * HEADS_;
  const long edgeTotal  = (long)BB * E * HEADS_;
  const long scatTotal  = edgeTotal * (FHEAD / 4);
  const long aggCount4  = (long)BB * NN * HIDD / 4;

  for (int layer = 0; layer < 2; ++layer) {
    const float* X    = layer ? AGG : feature;
    const int    K    = layer ? HIDD : DD;
    const float* W    = layer ? W2  : W1;
    const float* aS   = layer ? aS2 : aS1;
    const float* aD   = layer ? aD2 : aD1;
    const float* bias = layer ? b2  : b1;
    const float* g    = layer ? g2  : g1;
    const float* be   = layer ? be2 : be1;
    float* lnOut = layer ? (float*)d_out : AGG;   // layer1 in-place -> next input

    gemm_wmma<<<gemmBlocks, 256, 0, stream>>>(X, W, H, K, totalTiles, nodeTiles, colTiles);
    attn_logits<<<cdiv((long)segTotal * 32, 256), 256, 0, stream>>>(H, aS, aD, alS, alD, segTotal);
    init_seg<<<cdiv(segTotal, 256), 256, 0, stream>>>(mseg, denom, segTotal);
    zero_f4<<<cdiv(aggCount4, 256), 256, 0, stream>>>((float4*)AGG, aggCount4);
    edge_max_k<<<cdiv(edgeTotal, 256), 256, 0, stream>>>(src, dst, alS, alD, mseg, E, edgeTotal);
    edge_exp_k<<<cdiv(edgeTotal, 256), 256, 0, stream>>>(src, dst, alS, alD, mseg, denom, exbuf, E, edgeTotal);
    scatter_k<<<cdiv(scatTotal, 256), 256, 0, stream>>>(src, dst, H, exbuf, denom, AGG, E, scatTotal);
    ln_relu_k<<<cdiv((long)BB * NN * 32, 256), 256, 0, stream>>>(AGG, bias, g, be, lnOut, BB * NN);
  }
}
